// VectorQuantizer_29154238005540
// MI455X (gfx1250) — matrix operand order repaired and checked by reading, not compiled, not run on hardware
//
#include <hip/hip_runtime.h>

// Problem sizes (fixed by the reference): B=16, T=2048, D=512, K=4096
#define M_TOTAL (16 * 2048)   // 32768 rows
#define D_DIM   512
#define K_CODES 4096

#define BM       64           // rows per block
#define BN       128          // codes per chunk
#define KSTEP    32           // WMMA K per step (bf16)
#define DPIECE   64           // staged codebook piece depth (2 ksteps)
#define NCHUNKS  (K_CODES / BN)    // 32
#define NPIECES  (D_DIM / DPIECE)  // 8

#define XS_STRIDE 520         // 512 + 8 pad (ushorts) -> row stride 1040B
#define CS_STRIDE 72          // 64 + 8 pad  (ushorts) -> row stride 144B

#define SMEM_BYTES (((BM * XS_STRIDE) + 2 * (BN * CS_STRIDE)) * 2 /*bf16*/ \
                    + BM * 4 * 4 /*redval f32*/ + BM * 4 * 4 /*redidx i32*/ \
                    + BM * 4 /*bestk*/)

// Workspace layout: [0, 16KB) cnorm f32[4096]; [16KB, 16KB+4MB) codebook bf16
#define WS_CNORM_BYTES (K_CODES * 4)
#define WS_CBBF_BYTES  ((size_t)K_CODES * D_DIM * 2)
#define WS_NEED        (WS_CNORM_BYTES + WS_CBBF_BYTES)

typedef __attribute__((ext_vector_type(16))) __bf16        v16bf;
typedef __attribute__((ext_vector_type(8)))  float          v8f;
typedef __attribute__((ext_vector_type(4)))  float          f32x4;
typedef __attribute__((ext_vector_type(4)))  unsigned int   u32x4;
typedef __attribute__((ext_vector_type(8)))  unsigned int   u32x8;
typedef __attribute__((ext_vector_type(4)))  unsigned short u16x4;
typedef __attribute__((ext_vector_type(8)))  unsigned short u16x8;

__device__ __forceinline__ unsigned short f2bf(float f) {
  unsigned u = __builtin_bit_cast(unsigned, f);
  u += 0x7FFFu + ((u >> 16) & 1u);      // round-to-nearest-even to bf16
  return (unsigned short)(u >> 16);
}

// Build a 16-element bf16 fragment from two 16B LDS chunks (ds_load_b128 x2).
__device__ __forceinline__ v16bf frag16(const unsigned short* p, int off_ushorts) {
  u32x4 lo = *(const u32x4*)p;
  u32x4 hi = *(const u32x4*)(p + off_ushorts);
  u32x8 c;
  c[0] = lo[0]; c[1] = lo[1]; c[2] = lo[2]; c[3] = lo[3];
  c[4] = hi[0]; c[5] = hi[1]; c[6] = hi[2]; c[7] = hi[3];
  return __builtin_bit_cast(v16bf, c);
}

// ---------------------------------------------------------------------------
// Kernel 0a: pre-convert codebook f32 -> bf16 (one-time; hot loop then has
// zero conversion VALU). 256 thr x 2048 blocks, 4 elements/thread.
// ---------------------------------------------------------------------------
__global__ void vq_cbconv_kernel(const float* __restrict__ cb,
                                 unsigned short* __restrict__ cbb) {
  const size_t i = ((size_t)blockIdx.x * 256 + threadIdx.x) * 4;
  f32x4 v = *(const f32x4*)(cb + i);
  u16x4 h; h[0] = f2bf(v[0]); h[1] = f2bf(v[1]); h[2] = f2bf(v[2]); h[3] = f2bf(v[3]);
  *(u16x4*)(cbb + i) = h;
}

// ---------------------------------------------------------------------------
// Kernel 0b: codebook squared norms  cnorm[k] = sum_d c[k][d]^2
// ---------------------------------------------------------------------------
__global__ void vq_cnorm_kernel(const float* __restrict__ cb,
                                float* __restrict__ cnorm) {
  const int k = blockIdx.x;           // one code per block, 64 threads
  const int t = threadIdx.x;
  const float* row = cb + (size_t)k * D_DIM;
  float s = 0.f;
#pragma unroll
  for (int i = 0; i < D_DIM / 64; ++i) {
    float v = row[t + i * 64];
    s += v * v;
  }
#pragma unroll
  for (int m = 16; m >= 1; m >>= 1) s += __shfl_xor(s, m, 32);
  __shared__ float part[2];
  if ((t & 31) == 0) part[t >> 5] = s;
  __syncthreads();
  if (t == 0) cnorm[k] = part[0] + part[1];
}

// ---------------------------------------------------------------------------
// Kernel 1: fused  scores (WMMA bf16 GEMM)  ->  running argmin  ->  gather
// Block: 256 threads (8 waves as 2(M) x 4(N)), tile M=64 rows x all K codes.
// PRECONV=true streams pre-converted bf16 codebook (no conversion in loop).
// ---------------------------------------------------------------------------
template <bool PRECONV>
__global__ __launch_bounds__(256)
void vq_main_kernel(const float* __restrict__ x,
                    const float* __restrict__ cb,          // f32 codebook
                    const unsigned short* __restrict__ cbb, // bf16 codebook (or null)
                    const float* __restrict__ cnorm,
                    float* __restrict__ out) {
  extern __shared__ __align__(16) unsigned char smem[];
  unsigned short* Xs  = (unsigned short*)smem;                 // [64][520]
  unsigned short* Cs0 = Xs + BM * XS_STRIDE;                   // [128][72]
  unsigned short* Cs1 = Cs0 + BN * CS_STRIDE;                  // [128][72]
  float* redval = (float*)(Cs1 + BN * CS_STRIDE);              // [64][4]
  int*   redidx = (int*)(redval + BM * 4);                     // [64][4]
  int*   bestk  = (int*)(redidx + BM * 4);                     // [64]

  const int tid   = threadIdx.x;
  const int lane  = tid & 31;
  const int wid   = tid >> 5;
  const int mw    = wid >> 2;        // 0..1  -> rows [mw*32, mw*32+32)
  const int nw    = wid & 3;         // 0..3  -> cols [nw*32, nw*32+32) in chunk
  const int half  = lane >> 4;       // lane half (K split)
  const int lrow  = lane & 15;       // A row / B column within a 16-tile
  const int mblock = blockIdx.x * BM;

  // ---- Stage X tile (f32 -> bf16) into LDS, resident for the whole block ----
  {
    const int r2 = tid >> 7;         // 0..1
    const int c4 = tid & 127;        // float4 column
#pragma unroll 4
    for (int pass = 0; pass < 32; ++pass) {
      const int row = pass * 2 + r2;
      f32x4 v = *(const f32x4*)(x + (size_t)(mblock + row) * D_DIM + c4 * 4);
      u16x4 h; h[0] = f2bf(v[0]); h[1] = f2bf(v[1]); h[2] = f2bf(v[2]); h[3] = f2bf(v[3]);
      *(u16x4*)&Xs[row * XS_STRIDE + c4 * 4] = h;
    }
  }
  __syncthreads();

  // Codebook piece staging thread mappings.
  const int crow = tid >> 4;         // f32 path: 16 threads/row, float4 col
  const int ccol = tid & 15;
  const int hrow = tid >> 3;         // bf16 path: 8 threads/row, 16B segment
  const int hseg = tid & 7;
  f32x4 cregf[8];
  u16x8 cregh[4];

  // Running per-lane argmin state in WMMA C/D layout:
  //   slot (mi, ni, v) holds row (mw*32 + mi*16 + v + 8*half), column lrow.
  float runv[2][2][8];
  int   runi[2][2][8];
#pragma unroll
  for (int a = 0; a < 2; ++a)
#pragma unroll
    for (int b = 0; b < 2; ++b)
#pragma unroll
      for (int v = 0; v < 8; ++v) { runv[a][b][v] = __builtin_inff(); runi[a][b][v] = 0; }

  v8f acc[2][2];

  for (int chunk = 0; chunk < NCHUNKS; ++chunk) {
    const int nbase = chunk * BN;

    // ---- piece prefetch (global -> regs) ----
    auto loadC = [&](int p) {
      if constexpr (PRECONV) {
#pragma unroll
        for (int q = 0; q < 4; ++q)
          cregh[q] = *(const u16x8*)(cbb + (size_t)(nbase + q * 32 + hrow) * D_DIM +
                                     p * DPIECE + hseg * 8);
      } else {
#pragma unroll
        for (int pp = 0; pp < 8; ++pp)
          cregf[pp] = *(const f32x4*)(cb + (size_t)(nbase + pp * 16 + crow) * D_DIM +
                                      p * DPIECE + ccol * 4);
      }
    };
    // ---- piece commit (regs -> LDS, bf16) ----
    auto storeC = [&](unsigned short* buf) {
      if constexpr (PRECONV) {
#pragma unroll
        for (int q = 0; q < 4; ++q)
          *(u16x8*)&buf[(q * 32 + hrow) * CS_STRIDE + hseg * 8] = cregh[q];
      } else {
#pragma unroll
        for (int pp = 0; pp < 8; ++pp) {
          f32x4 v = cregf[pp];
          u16x4 h; h[0] = f2bf(v[0]); h[1] = f2bf(v[1]); h[2] = f2bf(v[2]); h[3] = f2bf(v[3]);
          *(u16x4*)&buf[(pp * 16 + crow) * CS_STRIDE + ccol * 4] = h;
        }
      }
    };

    loadC(0);
    storeC(Cs0);
    __syncthreads();

#pragma unroll
    for (int a = 0; a < 2; ++a)
#pragma unroll
      for (int b = 0; b < 2; ++b) acc[a][b] = (v8f)0.f;

    for (int p = 0; p < NPIECES; ++p) {
      unsigned short* cur = (p & 1) ? Cs1 : Cs0;
      unsigned short* nxt = (p & 1) ? Cs0 : Cs1;

      if (p < NPIECES - 1) loadC(p + 1);   // prefetch next piece while computing

#pragma unroll
      for (int ks2 = 0; ks2 < 2; ++ks2) {
        const int ks   = p * 2 + ks2;
        const int dsub = ks2 * KSTEP;
        // A fragments (16x32 bf16): lane m=lrow; K groups at half*8 and half*8+16.
        v16bf af[2];
#pragma unroll
        for (int mi = 0; mi < 2; ++mi) {
          const unsigned short* ap =
              &Xs[(mw * 32 + mi * 16 + lrow) * XS_STRIDE + ks * KSTEP + half * 8];
          af[mi] = frag16(ap, 16);
        }
        // B fragments (32x16 bf16): lane n=lrow; K groups at half*8 and half*8+16.
        v16bf bfr[2];
#pragma unroll
        for (int ni = 0; ni < 2; ++ni) {
          const unsigned short* bp =
              &cur[(nw * 32 + ni * 16 + lrow) * CS_STRIDE + dsub + half * 8];
          bfr[ni] = frag16(bp, 16);
        }
#pragma unroll
        for (int mi = 0; mi < 2; ++mi)
#pragma unroll
          for (int ni = 0; ni < 2; ++ni)
            acc[mi][ni] = __builtin_amdgcn_wmma_f32_16x16x32_bf16(
                false, af[mi], false, bfr[ni], (short)0, acc[mi][ni], false, false);
      }
      __syncthreads();
      if (p < NPIECES - 1) {
        storeC(nxt);
        __syncthreads();
      }
    }

    // Fold this chunk's scores into the running per-lane argmin (no shuffles).
#pragma unroll
    for (int ni = 0; ni < 2; ++ni) {
      const int code = nbase + nw * 32 + ni * 16 + lrow;
      const float cn = cnorm[code];
#pragma unroll
      for (int mi = 0; mi < 2; ++mi)
#pragma unroll
        for (int v = 0; v < 8; ++v) {
          const float d2 = cn - 2.f * acc[mi][ni][v];
          if (d2 < runv[mi][ni][v]) { runv[mi][ni][v] = d2; runi[mi][ni][v] = code; }
        }
    }
  }

  // ---- Final reduction: ni fold, then 16-lane shuffle argmin, then LDS ----
#pragma unroll
  for (int mi = 0; mi < 2; ++mi)
#pragma unroll
    for (int v = 0; v < 8; ++v) {
      float bv = runv[mi][0][v]; int bi = runi[mi][0][v];
      {
        const float ov = runv[mi][1][v]; const int oi = runi[mi][1][v];
        if (ov < bv || (ov == bv && oi < bi)) { bv = ov; bi = oi; }
      }
#pragma unroll
      for (int m = 1; m <= 8; m <<= 1) {
        const float ov = __shfl_xor(bv, m, 32);
        const int   oi = __shfl_xor(bi, m, 32);
        if (ov < bv || (ov == bv && oi < bi)) { bv = ov; bi = oi; }
      }
      if (lrow == 0) {
        const int row = mw * 32 + mi * 16 + v + 8 * half;
        redval[row * 4 + nw] = bv;
        redidx[row * 4 + nw] = bi;
      }
    }
  __syncthreads();

  if (tid < BM) {
    float bv = redval[tid * 4]; int bi = redidx[tid * 4];
#pragma unroll
    for (int j = 1; j < 4; ++j) {
      const float ov = redval[tid * 4 + j]; const int oi = redidx[tid * 4 + j];
      if (ov < bv || (ov == bv && oi < bi)) { bv = ov; bi = oi; }
    }
    bestk[tid] = bi;
  }
  __syncthreads();

  // ---- Gather winning codebook rows (f32, L2-resident) to output ----
  {
    const int r2 = tid >> 7;
    const int c4 = tid & 127;
#pragma unroll 4
    for (int pass = 0; pass < 32; ++pass) {
      const int row = pass * 2 + r2;
      const int k = bestk[row];
      f32x4 v = *(const f32x4*)(cb + (size_t)k * D_DIM + c4 * 4);
      *(f32x4*)(out + (size_t)(mblock + row) * D_DIM + c4 * 4) = v;
    }
  }
}

// ---------------------------------------------------------------------------
extern "C" void kernel_launch(void* const* d_in, const int* in_sizes, int n_in,
                              void* d_out, int out_size, void* d_ws, size_t ws_size,
                              hipStream_t stream) {
  (void)in_sizes; (void)n_in; (void)out_size;
  const float* x  = (const float*)d_in[0];   // [B,T,D] f32
  const float* cb = (const float*)d_in[1];   // [K,D]  f32
  float* out   = (float*)d_out;              // [B,T,D] f32
  float* cnorm = (float*)d_ws;               // K floats scratch
  unsigned short* cbb =
      (unsigned short*)((char*)d_ws + WS_CNORM_BYTES); // bf16 codebook scratch

  vq_cnorm_kernel<<<K_CODES, 64, 0, stream>>>(cb, cnorm);

  if (ws_size >= WS_NEED) {
    (void)hipFuncSetAttribute((const void*)vq_main_kernel<true>,
                              hipFuncAttributeMaxDynamicSharedMemorySize,
                              SMEM_BYTES);
    vq_cbconv_kernel<<<(K_CODES * D_DIM) / (256 * 4), 256, 0, stream>>>(cb, cbb);
    vq_main_kernel<true><<<M_TOTAL / BM, 256, SMEM_BYTES, stream>>>(
        x, cb, cbb, cnorm, out);
  } else {
    (void)hipFuncSetAttribute((const void*)vq_main_kernel<false>,
                              hipFuncAttributeMaxDynamicSharedMemorySize,
                              SMEM_BYTES);
    vq_main_kernel<false><<<M_TOTAL / BM, 256, SMEM_BYTES, stream>>>(
        x, cb, nullptr, cnorm, out);
  }
}
